// MambaForecaster_61486751810088
// MI455X (gfx1250) — compile-verified, matrix operand order checked by reading, not run on hardware
//
#include <hip/hip_runtime.h>

#define D_MODEL   256
#define N_LAYERS  4
#define CTX       512
#define PRED      96
#define D_STATE   32
#define D_CONV    4
#define D_INNER   512
#define NHEADS    8
#define HEADDIM   64
#define D_IN_PROJ 1096
#define CONV_DIM  576
#define BATCH     8
#define NTOK      (BATCH * CTX)   // 4096
#define EPS       1e-5f

typedef __attribute__((ext_vector_type(16))) _Float16 v16h;
typedef __attribute__((ext_vector_type(8)))  float    v8f;

__device__ __forceinline__ float silu_f(float x) { return x / (1.f + __expf(-x)); }

// ---------------------------------------------------------------------------
// f32 -> f16 operand pre-conversion (amortized across all tile reuse)
// ---------------------------------------------------------------------------
__global__ void f32_to_f16_kernel(const float* __restrict__ src,
                                  _Float16* __restrict__ dst, long n) {
  long i = (long)blockIdx.x * blockDim.x + threadIdx.x;
  if (i < n) dst[i] = (_Float16)src[i];
}

// Fragment loader per ISA 7.12.2 (16-bit A/B 16x32 layout):
// lane row = lane%16; halves 0-7 -> K {k..k+7}, halves 8-15 -> K {k+16..k+23};
// lanes 16-31 use k+8 (folded by caller). Rows pre-clamped -> always in bounds.
__device__ __forceinline__ v16h load_frag_h(const _Float16* __restrict__ p,
                                            int ld, int row, int k) {
  v16h out;
  const _Float16* q = p + (long)row * ld + k;
#pragma unroll
  for (int j = 0; j < 8; ++j) out[j] = q[j];
#pragma unroll
  for (int j = 0; j < 8; ++j) out[8 + j] = q[16 + j];
  return out;
}

// C[M,N] = A[M,K] * B[N,K]^T, f16 operands, f32 accumulate.
// One wave computes a 16x64 strip: 1 A-fragment reused by 4 independent WMMAs
// per K-step (hides the 5-slot WMMA->WMMA hazard). Row indices are clamped
// instead of branch-guarded: OOB rows only feed outputs that are never stored,
// and EXEC stays all-1s through the WMMAs.
__global__ void __launch_bounds__(256)
wmma_gemm_nt(const _Float16* __restrict__ A, const _Float16* __restrict__ B,
             float* __restrict__ C, int M, int N, int K, int ncg, int ntm) {
  int gw = (int)((blockIdx.x * blockDim.x + threadIdx.x) >> 5);
  int tm = gw / ncg;
  if (tm >= ntm) return;                    // wave-uniform exit
  int cg   = gw - tm * ncg;                 // 64-column group
  int lane = threadIdx.x & 31;
  int lrow = lane & 15;
  int koff = (lane >> 4) << 3;
  int arow  = min(tm * 16 + lrow, M - 1);
  int brow0 = cg * 64 + lrow;
  int b0 = min(brow0,      N - 1);
  int b1 = min(brow0 + 16, N - 1);
  int b2 = min(brow0 + 32, N - 1);
  int b3 = min(brow0 + 48, N - 1);
  v8f acc0 = {0.f,0.f,0.f,0.f,0.f,0.f,0.f,0.f};
  v8f acc1 = acc0, acc2 = acc0, acc3 = acc0;
  for (int k0 = 0; k0 < K; k0 += 32) {
    int k = k0 + koff;
    v16h a  = load_frag_h(A, K, arow, k);
    v16h f0 = load_frag_h(B, K, b0, k);
    v16h f1 = load_frag_h(B, K, b1, k);
    v16h f2 = load_frag_h(B, K, b2, k);
    v16h f3 = load_frag_h(B, K, b3, k);
    acc0 = __builtin_amdgcn_wmma_f32_16x16x32_f16(false, a, false, f0, (short)0, acc0, false, false);
    acc1 = __builtin_amdgcn_wmma_f32_16x16x32_f16(false, a, false, f1, (short)0, acc1, false, false);
    acc2 = __builtin_amdgcn_wmma_f32_16x16x32_f16(false, a, false, f2, (short)0, acc2, false, false);
    acc3 = __builtin_amdgcn_wmma_f32_16x16x32_f16(false, a, false, f3, (short)0, acc3, false, false);
  }
  // C/D layout: VGPR r -> M = r + 8*(lane/16), N = lane%16
  int mb = tm * 16 + ((lane >> 4) << 3);
#pragma unroll
  for (int t = 0; t < 4; ++t) {
    v8f acc = (t == 0) ? acc0 : (t == 1) ? acc1 : (t == 2) ? acc2 : acc3;
    int n = cg * 64 + t * 16 + lrow;
    if (n < N) {
#pragma unroll
      for (int r = 0; r < 8; ++r) {
        int m = mb + r;
        if (m < M) C[(long)m * N + n] = acc[r];
      }
    }
  }
}

// Per-(b,d) normalization over the time axis: mean / std(ddof=1), clip, write h.
__global__ void norm_time_kernel(const float* __restrict__ x, float* __restrict__ h,
                                 float* __restrict__ mean, float* __restrict__ stdv) {
  int id = blockIdx.x * blockDim.x + threadIdx.x;
  if (id >= BATCH * D_MODEL) return;
  int b = id / D_MODEL, d = id % D_MODEL;
  const float* xb = x + (long)b * CTX * D_MODEL + d;
  float s = 0.f;
  for (int l = 0; l < CTX; ++l) s += xb[(long)l * D_MODEL];
  float mu = s / (float)CTX;
  float ss = 0.f;
  for (int l = 0; l < CTX; ++l) { float dd = xb[(long)l * D_MODEL] - mu; ss += dd * dd; }
  float sd = sqrtf(ss / (float)(CTX - 1));
  sd = fmaxf(sd, 1e-4f) + 1e-5f;
  mean[id] = mu; stdv[id] = sd;
  float inv = 1.f / sd;
  float* hb = h + (long)b * CTX * D_MODEL + d;
  for (int l = 0; l < CTX; ++l) hb[(long)l * D_MODEL] = (xb[(long)l * D_MODEL] - mu) * inv;
}

// Depthwise causal conv (K=4) + bias + silu on xBC channels; softplus(dt) fused.
__global__ void conv_dt_kernel(const float* __restrict__ zx, const float* __restrict__ cw,
                               const float* __restrict__ cb, const float* __restrict__ dtb,
                               float* __restrict__ convout, float* __restrict__ dtout) {
  long idx = (long)blockIdx.x * blockDim.x + threadIdx.x;
  const long total = (long)NTOK * (CONV_DIM + NHEADS);
  if (idx >= total) return;
  int t = (int)(idx / (CONV_DIM + NHEADS));
  int c = (int)(idx % (CONV_DIM + NHEADS));
  int l = t & (CTX - 1);
  if (c < CONV_DIM) {
    float acc = cb[c];
#pragma unroll
    for (int k = 0; k < D_CONV; ++k) {
      int ll = l - (D_CONV - 1) + k;
      if (ll >= 0)
        acc += zx[(long)(t - l + ll) * D_IN_PROJ + D_INNER + c] * cw[c * D_CONV + k];
    }
    convout[(long)t * CONV_DIM + c] = silu_f(acc);
  } else {
    int hh = c - CONV_DIM;
    float v = zx[(long)t * D_IN_PROJ + (D_IN_PROJ - NHEADS) + hh] + dtb[hh];
    dtout[(long)t * NHEADS + hh] = (v > 20.f) ? v : log1pf(__expf(v));
  }
}

// Sequential SSD scan: one block per (batch, head), thread p owns 32 f32 states
// in registers; B/C broadcast through LDS each step.
__global__ void __launch_bounds__(64)
ssd_scan_kernel(const float* __restrict__ conv, const float* __restrict__ dt,
                const float* __restrict__ Alog, const float* __restrict__ Dp,
                float* __restrict__ y) {
  int b  = blockIdx.x / NHEADS;
  int hh = blockIdx.x % NHEADS;
  int p  = threadIdx.x;                 // 0..63 = HEADDIM
  __shared__ float sB[D_STATE];
  __shared__ float sC[D_STATE];
  __shared__ float sdt;
  float s[D_STATE];
#pragma unroll
  for (int n = 0; n < D_STATE; ++n) s[n] = 0.f;
  float A  = -__expf(Alog[hh]);
  float Dv = Dp[hh];
  for (int l = 0; l < CTX; ++l) {
    long t = (long)b * CTX + l;
    if (p < D_STATE)      sB[p]           = conv[t * CONV_DIM + D_INNER + p];
    else                  sC[p - D_STATE] = conv[t * CONV_DIM + D_INNER + D_STATE + (p - D_STATE)];
    if (p == 0) sdt = dt[t * NHEADS + hh];
    __syncthreads();
    float dtv  = sdt;
    float dA   = __expf(dtv * A);
    float xv   = conv[t * CONV_DIM + hh * HEADDIM + p];
    float coef = dtv * xv;
    float acc  = 0.f;
#pragma unroll
    for (int n = 0; n < D_STATE; ++n) {
      s[n] = s[n] * dA + coef * sB[n];
      acc += s[n] * sC[n];
    }
    y[t * D_INNER + hh * HEADDIM + p] = acc + Dv * xv;
    __syncthreads();
  }
}

// y *= silu(z); RMS-norm over D_INNER; * rms_w.  One block per token.
__global__ void __launch_bounds__(256)
gate_rms_kernel(float* __restrict__ y, const float* __restrict__ zx,
                const float* __restrict__ rw) {
  int t = blockIdx.x;
  int c = threadIdx.x;
  __shared__ float red[256];
  float y0 = y[(long)t * D_INNER + c]       * silu_f(zx[(long)t * D_IN_PROJ + c]);
  float y1 = y[(long)t * D_INNER + c + 256] * silu_f(zx[(long)t * D_IN_PROJ + c + 256]);
  red[c] = y0 * y0 + y1 * y1;
  __syncthreads();
  for (int s = 128; s > 0; s >>= 1) {
    if (c < s) red[c] += red[c + s];
    __syncthreads();
  }
  float scale = rsqrtf(red[0] / (float)D_INNER + EPS);
  y[(long)t * D_INNER + c]       = y0 * scale * rw[c];
  y[(long)t * D_INNER + c + 256] = y1 * scale * rw[c + 256];
}

// h = LayerNorm(h + block_out) * g + b.  One block per token.
__global__ void __launch_bounds__(256)
res_ln_kernel(float* __restrict__ h, const float* __restrict__ bo,
              const float* __restrict__ g, const float* __restrict__ bt) {
  int t = blockIdx.x;
  int d = threadIdx.x;
  __shared__ float rsum[256];
  __shared__ float rsq[256];
  float v = h[(long)t * D_MODEL + d] + bo[(long)t * D_MODEL + d];
  rsum[d] = v; rsq[d] = v * v;
  __syncthreads();
  for (int s = 128; s > 0; s >>= 1) {
    if (d < s) { rsum[d] += rsum[d + s]; rsq[d] += rsq[d + s]; }
    __syncthreads();
  }
  float mu  = rsum[0] / (float)D_MODEL;
  float var = rsq[0] / (float)D_MODEL - mu * mu;
  h[(long)t * D_MODEL + d] = (v - mu) * rsqrtf(var + EPS) * g[d] + bt[d];
}

__global__ void gather_last_kernel(const float* __restrict__ h, float* __restrict__ last) {
  int id = blockIdx.x * blockDim.x + threadIdx.x;
  if (id >= BATCH * D_MODEL) return;
  int b = id / D_MODEL, d = id % D_MODEL;
  last[id] = h[((long)b * CTX + (CTX - 1)) * D_MODEL + d];
}

__global__ void head_epilogue_kernel(const float* __restrict__ pr, const float* __restrict__ hb,
                                     const float* __restrict__ mean, const float* __restrict__ stdv,
                                     float* __restrict__ out) {
  long idx = (long)blockIdx.x * blockDim.x + threadIdx.x;
  const long NOUT = (long)PRED * D_MODEL;     // 24576
  if (idx >= BATCH * NOUT) return;
  int b = (int)(idx / NOUT);
  int n = (int)(idx % NOUT);
  int d = n & (D_MODEL - 1);
  out[idx] = (pr[idx] + hb[n]) * stdv[b * D_MODEL + d] + mean[b * D_MODEL + d];
}

static inline int cdiv_blocks(long n) { return (int)((n + 255) / 256); }

extern "C" void kernel_launch(void* const* d_in, const int* in_sizes, int n_in,
                              void* d_out, int out_size, void* d_ws, size_t ws_size,
                              hipStream_t stream) {
  const float* x       = (const float*)d_in[0];
  const float* Wi_all  = (const float*)d_in[1];
  const float* cw_all  = (const float*)d_in[2];
  const float* cb_all  = (const float*)d_in[3];
  const float* dtb_all = (const float*)d_in[4];
  const float* Al_all  = (const float*)d_in[5];
  const float* Dp_all  = (const float*)d_in[6];
  const float* rw_all  = (const float*)d_in[7];
  const float* Wo_all  = (const float*)d_in[8];
  const float* lg_all  = (const float*)d_in[9];
  const float* lb_all  = (const float*)d_in[10];
  const float* hw      = (const float*)d_in[11];
  const float* hbias   = (const float*)d_in[12];
  float* out = (float*)d_out;

  // ---- scratch carve-out (f32 region, then 16B-aligned f16 region) ----
  float* ws = (float*)d_ws;
  float* mean_b = ws;  ws += BATCH * D_MODEL;
  float* std_b  = ws;  ws += BATCH * D_MODEL;
  float* h      = ws;  ws += (long)NTOK * D_MODEL;
  float* zx     = ws;  ws += (long)NTOK * D_IN_PROJ;
  float* convo  = ws;  ws += (long)NTOK * CONV_DIM;
  float* dts    = ws;  ws += (long)NTOK * NHEADS;
  float* ybuf   = ws;  ws += (long)NTOK * D_INNER;
  float* blko   = ws;  ws += (long)NTOK * D_MODEL;
  float* lastb  = ws;  ws += BATCH * D_MODEL;
  float* predr  = ws;  ws += (long)BATCH * PRED * D_MODEL;

  _Float16* hp = (_Float16*)ws;
  _Float16* hH     = hp;  hp += (long)NTOK * D_MODEL;
  _Float16* ybufH  = hp;  hp += (long)NTOK * D_INNER;
  _Float16* lastbH = hp;  hp += BATCH * D_MODEL;
  _Float16* WiH    = hp;  hp += (long)N_LAYERS * D_IN_PROJ * D_MODEL;
  _Float16* WoH    = hp;  hp += (long)N_LAYERS * D_MODEL * D_INNER;
  _Float16* hwH    = hp;  hp += (long)PRED * D_MODEL * D_MODEL;

  // ---- one-time weight pre-conversion to f16 ----
  {
    long n = (long)N_LAYERS * D_IN_PROJ * D_MODEL;
    f32_to_f16_kernel<<<cdiv_blocks(n), 256, 0, stream>>>(Wi_all, WiH, n);
  }
  {
    long n = (long)N_LAYERS * D_MODEL * D_INNER;
    f32_to_f16_kernel<<<cdiv_blocks(n), 256, 0, stream>>>(Wo_all, WoH, n);
  }
  {
    long n = (long)PRED * D_MODEL * D_MODEL;
    f32_to_f16_kernel<<<cdiv_blocks(n), 256, 0, stream>>>(hw, hwH, n);
  }

  norm_time_kernel<<<cdiv_blocks(BATCH * D_MODEL), 256, 0, stream>>>(x, h, mean_b, std_b);

  for (int i = 0; i < N_LAYERS; ++i) {
    const float* cw  = cw_all  + (long)i * CONV_DIM * D_CONV;
    const float* cb  = cb_all  + (long)i * CONV_DIM;
    const float* dtb = dtb_all + (long)i * NHEADS;
    const float* Al  = Al_all  + (long)i * NHEADS;
    const float* Dpl = Dp_all  + (long)i * NHEADS;
    const float* rwl = rw_all  + (long)i * D_INNER;
    const float* lg  = lg_all  + (long)i * D_MODEL;
    const float* lb  = lb_all  + (long)i * D_MODEL;
    const _Float16* WiHl = WiH + (long)i * D_IN_PROJ * D_MODEL;
    const _Float16* WoHl = WoH + (long)i * D_MODEL * D_INNER;

    {   // h -> f16
      long n = (long)NTOK * D_MODEL;
      f32_to_f16_kernel<<<cdiv_blocks(n), 256, 0, stream>>>(h, hH, n);
    }
    {   // in-proj: zx[NTOK,1096] = h * Wi^T
      int ntm = NTOK / 16, ncg = (D_IN_PROJ + 63) / 64;
      int blocks = (int)(((long)ntm * ncg * 32 + 255) / 256);
      wmma_gemm_nt<<<blocks, 256, 0, stream>>>(hH, WiHl, zx, NTOK, D_IN_PROJ, D_MODEL, ncg, ntm);
    }
    {
      long total = (long)NTOK * (CONV_DIM + NHEADS);
      conv_dt_kernel<<<cdiv_blocks(total), 256, 0, stream>>>(zx, cw, cb, dtb, convo, dts);
    }
    ssd_scan_kernel<<<BATCH * NHEADS, HEADDIM, 0, stream>>>(convo, dts, Al, Dpl, ybuf);
    gate_rms_kernel<<<NTOK, 256, 0, stream>>>(ybuf, zx, rwl);
    {   // ybuf -> f16
      long n = (long)NTOK * D_INNER;
      f32_to_f16_kernel<<<cdiv_blocks(n), 256, 0, stream>>>(ybuf, ybufH, n);
    }
    {   // out-proj: blko[NTOK,256] = ybuf * Wo^T
      int ntm = NTOK / 16, ncg = D_MODEL / 64;
      int blocks = (int)(((long)ntm * ncg * 32 + 255) / 256);
      wmma_gemm_nt<<<blocks, 256, 0, stream>>>(ybufH, WoHl, blko, NTOK, D_MODEL, D_INNER, ncg, ntm);
    }
    res_ln_kernel<<<NTOK, 256, 0, stream>>>(h, blko, lg, lb);
  }

  gather_last_kernel<<<cdiv_blocks(BATCH * D_MODEL), 256, 0, stream>>>(h, lastb);
  {
    long n = BATCH * D_MODEL;
    f32_to_f16_kernel<<<cdiv_blocks(n), 256, 0, stream>>>(lastb, lastbH, n);
  }
  {   // head: predr[8,24576] = lastb * hw^T  (M padded to 16 in-tile)
    int ntm = 1, ncg = (PRED * D_MODEL) / 64;
    int blocks = (int)(((long)ntm * ncg * 32 + 255) / 256);
    wmma_gemm_nt<<<blocks, 256, 0, stream>>>(lastbH, hwH, predr, BATCH, PRED * D_MODEL, D_MODEL, ncg, ntm);
  }
  {
    long total = (long)BATCH * PRED * D_MODEL;
    head_epilogue_kernel<<<cdiv_blocks(total), 256, 0, stream>>>(predr, hbias, mean_b, std_b, out);
  }
}
